// GATModel_59330678226974
// MI455X (gfx1250) — matrix operand order repaired and checked by reading, not compiled, not run on hardware
//
#include <hip/hip_runtime.h>
#include <hip/hip_bf16.h>

typedef __attribute__((ext_vector_type(16))) __bf16 v16bf;
typedef __attribute__((ext_vector_type(8)))  float  v8f;

#define FIN 128
#define HEADS 8
#define FOUT 16
#define NEG_SLOPE 0.2f

// ---------------------------------------------------------------------------
// order-preserving float <-> uint mapping for atomicMax on floats
// ---------------------------------------------------------------------------
__device__ __forceinline__ unsigned f2key(float f) {
    unsigned u = __float_as_uint(f);
    return (u & 0x80000000u) ? ~u : (u | 0x80000000u);
}
__device__ __forceinline__ float key2f(unsigned k) {
    unsigned u = (k & 0x80000000u) ? (k & 0x7fffffffu) : ~k;
    return __uint_as_float(u);
}

// ---------------------------------------------------------------------------
// Init: zero output accumulator, denominators; bottom-out the max keys
// ---------------------------------------------------------------------------
__global__ void gat_init(float* __restrict__ out, float* __restrict__ denom,
                         unsigned* __restrict__ emax, int n) {
    long long i = (long long)blockIdx.x * blockDim.x + threadIdx.x;
    long long tot = (long long)n * FIN;
    if (i < tot) out[i] = 0.0f;
    if (i < (long long)n * HEADS) {
        denom[i] = 0.0f;
        emax[i]  = 0u;     // key space bottom (maps below every real float)
    }
}

// ---------------------------------------------------------------------------
// Projection: xp = x @ W  (bf16 WMMA, f32 accum), fused a_src/a_dst dots.
// Block = 128 threads = 4 waves; each wave computes a 16x128 panel.
// ---------------------------------------------------------------------------
__global__ __launch_bounds__(128) void gat_proj(
    const float* __restrict__ x, const float* __restrict__ W,
    const float* __restrict__ att_src, const float* __restrict__ att_dst,
    float* __restrict__ xp, float* __restrict__ asrc, float* __restrict__ adst,
    int n_nodes) {

    __shared__ __bf16 Wt[FIN][136];   // transposed W, padded pitch (bank-friendly)

    const int tid = threadIdx.x;
    // Stage Wt[n][k] = W[k][n] as bf16 (W fits in L2; one 64KB read per block)
    {
        const int n = tid;                     // 0..127 (column of W)
        for (int k = 0; k < FIN; ++k)
            Wt[n][k] = (__bf16)W[k * FIN + n];
    }
    __syncthreads();

    const int wave = tid >> 5;
    const int lane = tid & 31;
    const int hh   = lane >> 4;        // lane half (0/1)
    const int ln   = lane & 15;
    const int row0 = blockIdx.x * 64 + wave * 16;

    // clamped row pointer so EXEC stays all-ones through the WMMA loop
    int mrow  = row0 + ln;
    int mload = (mrow < n_nodes) ? mrow : (n_nodes - 1);
    const float* xrow = x + (size_t)mload * FIN;

    v8f acc[8];
    for (int nt = 0; nt < 8; ++nt)
        for (int r = 0; r < 8; ++r) acc[nt][r] = 0.0f;

    for (int kt = 0; kt < 4; ++kt) {
        // A fragment: lane half 0 -> K {kt*32+0..7, kt*32+16..23}; half 1 -> +8
        const float* a0 = xrow + kt * 32 + hh * 8;
        const float* a1 = a0 + 16;
        float4 f0 = ((const float4*)a0)[0];
        float4 f1 = ((const float4*)a0)[1];
        float4 f2 = ((const float4*)a1)[0];
        float4 f3 = ((const float4*)a1)[1];
        v16bf afrag;
        afrag[0]  = (__bf16)f0.x; afrag[1]  = (__bf16)f0.y;
        afrag[2]  = (__bf16)f0.z; afrag[3]  = (__bf16)f0.w;
        afrag[4]  = (__bf16)f1.x; afrag[5]  = (__bf16)f1.y;
        afrag[6]  = (__bf16)f1.z; afrag[7]  = (__bf16)f1.w;
        afrag[8]  = (__bf16)f2.x; afrag[9]  = (__bf16)f2.y;
        afrag[10] = (__bf16)f2.z; afrag[11] = (__bf16)f2.w;
        afrag[12] = (__bf16)f3.x; afrag[13] = (__bf16)f3.y;
        afrag[14] = (__bf16)f3.z; afrag[15] = (__bf16)f3.w;

        for (int nt = 0; nt < 8; ++nt) {
            // B fragment: column n = nt*16+ln, K {kt*32+hh*8+0..7, kt*32+16+hh*8+0..7}
            const __bf16* bp = &Wt[nt * 16 + ln][kt * 32 + hh * 8];
            v16bf bfrag;
            ((float4*)&bfrag)[0] = *(const float4*)(bp);
            ((float4*)&bfrag)[1] = *(const float4*)(bp + 16);
            acc[nt] = __builtin_amdgcn_wmma_f32_16x16x32_bf16(
                false, afrag, false, bfrag, (short)0, acc[nt], false, false);
        }
    }

    // Store xp: lane holds D rows m = hh*8 + r for column n = ln of tile nt
    for (int nt = 0; nt < 8; ++nt) {
        for (int r = 0; r < 8; ++r) {
            int m = row0 + hh * 8 + r;
            if (m < n_nodes)
                xp[(size_t)m * FIN + nt * 16 + ln] = acc[nt][r];
        }
    }

    // Fused attention dots: tile nt == head nt (FOUT==16)
    for (int nt = 0; nt < 8; ++nt) {
        float cs = att_src[nt * 16 + ln];
        float cd = att_dst[nt * 16 + ln];
        float ps[8], pd[8];
        for (int r = 0; r < 8; ++r) {
            ps[r] = acc[nt][r] * cs;
            pd[r] = acc[nt][r] * cd;
        }
        for (int s = 1; s <= 8; s <<= 1) {     // reduce over ln within half
            for (int r = 0; r < 8; ++r) {
                ps[r] += __shfl_xor(ps[r], s, 32);
                pd[r] += __shfl_xor(pd[r], s, 32);
            }
        }
        if (ln < 8) {                           // lane ln writes r = ln
            int m = row0 + hh * 8 + ln;
            if (m < n_nodes) {
                asrc[(size_t)m * HEADS + nt] = ps[ln];
                adst[(size_t)m * HEADS + nt] = pd[ln];
            }
        }
    }
}

// ---------------------------------------------------------------------------
// Edge helpers: one thread per (edge, head). Edges >= E are self loops.
// ---------------------------------------------------------------------------
__device__ __forceinline__ void edge_sd(const long long* __restrict__ ei,
                                        long long e, int E, int* s, int* d) {
    if (e < E) { *s = (int)ei[e]; *d = (int)ei[E + e]; }
    else       { *s = *d = (int)(e - E); }
}

__global__ void gat_edge_max(const long long* __restrict__ ei,
                             const float* __restrict__ asrc,
                             const float* __restrict__ adst,
                             unsigned* __restrict__ emax, int E, int n) {
    long long t = (long long)blockIdx.x * blockDim.x + threadIdx.x;
    long long total = (long long)(E + n) * HEADS;
    if (t >= total) return;
    int h = (int)(t & 7);
    long long e = t >> 3;
    int s, d; edge_sd(ei, e, E, &s, &d);
    float v = asrc[(size_t)s * HEADS + h] + adst[(size_t)d * HEADS + h];
    v = (v > 0.0f) ? v : v * NEG_SLOPE;
    atomicMax(&emax[(size_t)d * HEADS + h], f2key(v));
}

__global__ void gat_edge_denom(const long long* __restrict__ ei,
                               const float* __restrict__ asrc,
                               const float* __restrict__ adst,
                               const unsigned* __restrict__ emax,
                               float* __restrict__ denom, int E, int n) {
    long long t = (long long)blockIdx.x * blockDim.x + threadIdx.x;
    long long total = (long long)(E + n) * HEADS;
    if (t >= total) return;
    int h = (int)(t & 7);
    long long e = t >> 3;
    int s, d; edge_sd(ei, e, E, &s, &d);
    float v = asrc[(size_t)s * HEADS + h] + adst[(size_t)d * HEADS + h];
    v = (v > 0.0f) ? v : v * NEG_SLOPE;
    float m = key2f(emax[(size_t)d * HEADS + h]);
    atomicAdd(&denom[(size_t)d * HEADS + h], __expf(v - m));
}

__global__ void gat_edge_aggr(const long long* __restrict__ ei,
                              const float* __restrict__ asrc,
                              const float* __restrict__ adst,
                              const unsigned* __restrict__ emax,
                              const float* __restrict__ denom,
                              const float* __restrict__ xp,
                              float* __restrict__ out, int E, int n) {
    long long t = (long long)blockIdx.x * blockDim.x + threadIdx.x;
    long long total = (long long)(E + n) * HEADS;
    if (t >= total) return;
    int h = (int)(t & 7);
    long long e = t >> 3;
    int s, d; edge_sd(ei, e, E, &s, &d);
    float v = asrc[(size_t)s * HEADS + h] + adst[(size_t)d * HEADS + h];
    v = (v > 0.0f) ? v : v * NEG_SLOPE;
    float m = key2f(emax[(size_t)d * HEADS + h]);
    float alpha = __expf(v - m) / (denom[(size_t)d * HEADS + h] + 1e-16f);

    const float4* xs = (const float4*)(xp + (size_t)s * FIN + h * FOUT);
    float* op = out + (size_t)d * FIN + h * FOUT;
    for (int j = 0; j < 4; ++j) {
        float4 q = xs[j];
        atomicAdd(op + j * 4 + 0, alpha * q.x);
        atomicAdd(op + j * 4 + 1, alpha * q.y);
        atomicAdd(op + j * 4 + 2, alpha * q.z);
        atomicAdd(op + j * 4 + 3, alpha * q.w);
    }
}

// ---------------------------------------------------------------------------
// Epilogue: ReLU + bias + log_softmax over the 128 features (warp per node)
// ---------------------------------------------------------------------------
__global__ __launch_bounds__(256) void gat_finish(float* __restrict__ out,
                                                  const float* __restrict__ bias,
                                                  int n) {
    int node = blockIdx.x * 8 + (threadIdx.x >> 5);
    int lane = threadIdx.x & 31;
    if (node >= n) return;
    float* row = out + (size_t)node * FIN;
    float v[4];
    for (int j = 0; j < 4; ++j) {
        float t = row[lane + 32 * j] + bias[lane + 32 * j];
        v[j] = (t > 0.0f) ? t : 0.0f;
    }
    float mx = fmaxf(fmaxf(v[0], v[1]), fmaxf(v[2], v[3]));
    for (int s = 1; s < 32; s <<= 1) mx = fmaxf(mx, __shfl_xor(mx, s, 32));
    float se = 0.0f;
    for (int j = 0; j < 4; ++j) se += __expf(v[j] - mx);
    for (int s = 1; s < 32; s <<= 1) se += __shfl_xor(se, s, 32);
    float lse = mx + __logf(se);
    for (int j = 0; j < 4; ++j) row[lane + 32 * j] = v[j] - lse;
}

// ---------------------------------------------------------------------------
extern "C" void kernel_launch(void* const* d_in, const int* in_sizes, int n_in,
                              void* d_out, int out_size, void* d_ws, size_t ws_size,
                              hipStream_t stream) {
    const float*     x    = (const float*)d_in[0];
    const long long* ei   = (const long long*)d_in[1];
    const float*     W    = (const float*)d_in[2];
    const float*     aS   = (const float*)d_in[3];
    const float*     aD   = (const float*)d_in[4];
    const float*     bias = (const float*)d_in[5];
    float*           out  = (float*)d_out;

    const int N = in_sizes[0] / FIN;      // 100000
    const int E = in_sizes[1] / 2;        // 1600000

    // workspace layout (all L2-resident reduction state + xp)
    float*    xp    = (float*)d_ws;
    float*    asrc  = xp   + (size_t)N * FIN;
    float*    adst  = asrc + (size_t)N * HEADS;
    unsigned* emax  = (unsigned*)(adst + (size_t)N * HEADS);
    float*    denom = (float*)(emax + (size_t)N * HEADS);

    {   // init accumulators
        long long tot = (long long)N * FIN;
        int blocks = (int)((tot + 255) / 256);
        gat_init<<<blocks, 256, 0, stream>>>(out, denom, emax, N);
    }
    {   // WMMA projection + fused attention dots
        int blocks = (N + 63) / 64;
        gat_proj<<<blocks, 128, 0, stream>>>(x, W, aS, aD, xp, asrc, adst, N);
    }
    {   // edge phase: max -> denom -> weighted scatter
        long long work = (long long)(E + N) * HEADS;
        int blocks = (int)((work + 255) / 256);
        gat_edge_max  <<<blocks, 256, 0, stream>>>(ei, asrc, adst, emax, E, N);
        gat_edge_denom<<<blocks, 256, 0, stream>>>(ei, asrc, adst, emax, denom, E, N);
        gat_edge_aggr <<<blocks, 256, 0, stream>>>(ei, asrc, adst, emax, denom, xp, out, E, N);
    }
    {   // ReLU + bias + log_softmax (warp per node, in place on d_out)
        int blocks = (N + 7) / 8;
        gat_finish<<<blocks, 256, 0, stream>>>(out, bias, N);
    }
}
